// SelfAttention_83803401879905
// MI455X (gfx1250) — compile-verified, matrix operand order checked by reading, not run on hardware
//
#include <hip/hip_runtime.h>

#define B_   2
#define S_   2048
#define E_   5120
#define HQ_  32
#define HKV_ 8
#define HD_  128
#define DQ_  (HQ_ * HD_)    // 4096
#define DKV_ (HKV_ * HD_)   // 1024
#define NEGINF (-1.0e30f)

typedef _Float16 v16h __attribute__((ext_vector_type(16)));
typedef _Float16 v8h  __attribute__((ext_vector_type(8)));
typedef float    v8f  __attribute__((ext_vector_type(8)));
typedef int      i4v  __attribute__((__vector_size__(4 * sizeof(int))));

// CDNA5 async global->LDS path (ASYNCcnt). Guarded: falls back to direct loads.
#if defined(__gfx1250__) && __has_builtin(__builtin_amdgcn_global_load_async_to_lds_b128) && \
    __has_builtin(__builtin_amdgcn_s_wait_asynccnt)
#define USE_ASYNC_LDS 1
#else
#define USE_ASYNC_LDS 0
#endif

// Load one 16-half WMMA fragment chunk: halves [0..7] at p, [8..15] at p+16.
// CDNA5 16-bit A/B layout: lanes 0-15 hold K {0-7,16-23}, lanes 16-31 hold
// K {8-15,24-31}; caller passes p already offset by +8*half.
static __device__ __forceinline__ v16h frag_ld(const _Float16* p) {
  v8h lo = *(const v8h*)(p);
  v8h hi = *(const v8h*)(p + 16);
  v16h r;
#pragma unroll
  for (int i = 0; i < 8; ++i) { r[i] = lo[i]; r[i + 8] = hi[i]; }
  return r;
}

#if USE_ASYNC_LDS
// Copy nbytes (multiple of 512) from global src to LDS dst, 16B per lane per op.
static __device__ __forceinline__ void async_copy(_Float16* dst, const _Float16* src,
                                                  int nbytes, int lane) {
  for (int off = lane * 16; off < nbytes; off += 32 * 16) {
    __builtin_amdgcn_global_load_async_to_lds_b128(
        (__attribute__((address_space(1))) i4v*)((const char*)src + off),
        (__attribute__((address_space(3))) i4v*)((char*)dst + off), 0, 0);
  }
}
#endif

// ---------------------------------------------------------------- convert f32->f16
__global__ void cvt_f32_f16(const float* __restrict__ in, _Float16* __restrict__ out, long n) {
  long i = (long)blockIdx.x * blockDim.x + threadIdx.x;
  long stride = (long)gridDim.x * blockDim.x;
  for (; i < n; i += stride) out[i] = (_Float16)in[i];
}

// ---------------------------------------------------------------- GEMM: C[M,N] = A[M,K] * W[N,K]^T
// 256 threads = 8 waves; wave grid 2x4; each wave: 32 rows x 64 cols = 8 WMMA tiles.
__global__ void __launch_bounds__(256) gemm_h_nt(const _Float16* __restrict__ A,
                                                 const _Float16* __restrict__ W,
                                                 float* __restrict__ C,
                                                 int M, int N, int K) {
  const int lane = threadIdx.x & 31;
  const int wave = threadIdx.x >> 5;
  const int half = lane >> 4;
  const int lr   = lane & 15;
  const int row0 = blockIdx.y * 64 + (wave >> 2) * 32;
  const int col0 = blockIdx.x * 256 + (wave & 3) * 64;

  v8f acc[2][4];
#pragma unroll
  for (int r = 0; r < 2; ++r)
#pragma unroll
    for (int t = 0; t < 4; ++t)
#pragma unroll
      for (int i = 0; i < 8; ++i) acc[r][t][i] = 0.0f;

  const _Float16* Ar0 = A + (size_t)(row0 + lr) * K + 8 * half;
  const _Float16* Ar1 = Ar0 + (size_t)16 * K;
  const _Float16* Wr[4];
#pragma unroll
  for (int t = 0; t < 4; ++t) Wr[t] = W + (size_t)(col0 + t * 16 + lr) * K + 8 * half;

  for (int k0 = 0; k0 < K; k0 += 32) {
    if (k0 + 128 < K) {   // stream-ahead hint -> global_prefetch_b8
      __builtin_prefetch(Ar0 + k0 + 128, 0, 1);
      __builtin_prefetch(Ar1 + k0 + 128, 0, 1);
    }
    v16h a0 = frag_ld(Ar0 + k0);
    v16h a1 = frag_ld(Ar1 + k0);
#pragma unroll
    for (int t = 0; t < 4; ++t) {
      v16h b = frag_ld(Wr[t] + k0);
      acc[0][t] = __builtin_amdgcn_wmma_f32_16x16x32_f16(false, a0, false, b,
                                                         (short)0, acc[0][t], false, false);
      acc[1][t] = __builtin_amdgcn_wmma_f32_16x16x32_f16(false, a1, false, b,
                                                         (short)0, acc[1][t], false, false);
    }
  }
#pragma unroll
  for (int r = 0; r < 2; ++r)
#pragma unroll
    for (int t = 0; t < 4; ++t)
#pragma unroll
      for (int i = 0; i < 8; ++i)
        C[(size_t)(row0 + r * 16 + i + 8 * half) * N + col0 + t * 16 + lr] = acc[r][t][i];
}

// ---------------------------------------------------------------- RoPE + head-major f16 pack
__global__ void rope_pack(const float* __restrict__ X, const float* __restrict__ Cs,
                          const float* __restrict__ Sn, _Float16* __restrict__ out,
                          int H, long total) {
  long t = (long)blockIdx.x * blockDim.x + threadIdx.x;
  if (t >= total) return;                 // total = B*S*H*(HD/2)
  int j = (int)(t & 63); long u = t >> 6;
  int h = (int)(u % H);  u /= H;
  int s = (int)(u % S_);
  int b = (int)(u / S_);
  size_t ii = (size_t)(b * S_ + s) * ((size_t)H * HD_) + (size_t)h * HD_ + 2 * j;
  float xr = X[ii], xi = X[ii + 1];
  float c = Cs[s * 64 + j], sn = Sn[s * 64 + j];
  size_t oo = (((size_t)b * H + h) * S_ + s) * HD_ + 2 * j;
  out[oo]     = (_Float16)(xr * c - xi * sn);
  out[oo + 1] = (_Float16)(xr * sn + xi * c);
}

// V: [B*S, DKV] f32 -> Vt: [B, HKV, HD, S] f16 (transposed: P*V B-frags contiguous)
__global__ void v_pack(const float* __restrict__ V, _Float16* __restrict__ Vt, long total) {
  long t = (long)blockIdx.x * blockDim.x + threadIdx.x;
  if (t >= total) return;                 // total = B*S*HKV*HD
  int d = (int)(t % HD_); long u = t / HD_;
  int h = (int)(u % HKV_); u /= HKV_;
  int s = (int)(u % S_);
  int b = (int)(u / S_);
  Vt[((size_t)(b * HKV_ + h) * HD_ + d) * S_ + s] =
      (_Float16)V[(size_t)(b * S_ + s) * DKV_ + (size_t)h * HD_ + d];
}

// ---------------------------------------------------------------- Flash attention
// 128 threads = 4 independent waves; one wave per (b, head, 16-query tile).
// K tiles double-buffered in LDS via async copies (overlap with WMMA) when available.
__global__ void __launch_bounds__(128) flash_attn(const _Float16* __restrict__ Q,
                                                  const _Float16* __restrict__ K,
                                                  const _Float16* __restrict__ Vt,
                                                  _Float16* __restrict__ O) {
  const int lane = threadIdx.x & 31;
  const int wave = threadIdx.x >> 5;
  const int half = lane >> 4;
  const int lr   = lane & 15;
  const int qb   = (blockIdx.x * 4 + wave) * 16;
  const int h    = blockIdx.y;
  const int b    = blockIdx.z;
  const int hkv  = h / (HQ_ / HKV_);

  __shared__ __align__(16) _Float16 ldsP[4][16 * 32];
  _Float16* myP = ldsP[wave];
#if USE_ASYNC_LDS
  __shared__ __align__(16) _Float16 kbuf[4][2][32 * HD_];
  _Float16* kb0 = kbuf[wave][0];
  _Float16* kb1 = kbuf[wave][1];
#endif

  const _Float16* Qbase = Q  + (((size_t)b * HQ_ + h) * S_ + qb) * HD_;
  const _Float16* Kbase = K  + ((size_t)b * HKV_ + hkv) * (size_t)S_ * HD_;
  const _Float16* Vbase = Vt + ((size_t)b * HKV_ + hkv) * (size_t)HD_ * S_;

  v16h qf[4];
#pragma unroll
  for (int kk = 0; kk < 4; ++kk)
    qf[kk] = frag_ld(Qbase + (size_t)lr * HD_ + kk * 32 + 8 * half);

  v8f of[8];
  float m[8], l[8];
#pragma unroll
  for (int t = 0; t < 8; ++t) {
#pragma unroll
    for (int i = 0; i < 8; ++i) of[t][i] = 0.0f;
    m[t] = NEGINF; l[t] = 0.0f;
  }

  const float scale = 0.088388347648318447f;   // 1/sqrt(128)
  const int kend = qb + 16;                    // causal: keys [0, qb+16)

#if USE_ASYNC_LDS
  {
    const int nrows0 = (16 < kend) ? 32 : 16;
    async_copy(kb0, Kbase, nrows0 * HD_ * 2, lane);
    __builtin_amdgcn_s_wait_asynccnt(0);
  }
  int cur = 0;
#endif

  for (int kb = 0; kb < kend; kb += 32) {
    const bool do1 = (kb + 16) < kend;
#if USE_ASYNC_LDS
    const int nkb = kb + 32;
    if (nkb < kend) {          // prefetch next K tile while we compute this one
      const int nrows = ((nkb + 16) < kend) ? 32 : 16;
      async_copy(cur ? kb0 : kb1, Kbase + (size_t)nkb * HD_, nrows * HD_ * 2, lane);
    }
    const _Float16* Kt = cur ? kb1 : kb0;
#endif

    // ---- scores for keys [kb, kb+16) and [kb+16, kb+32)
    v8f sa, sb;
#pragma unroll
    for (int i = 0; i < 8; ++i) { sa[i] = 0.0f; sb[i] = 0.0f; }
#pragma unroll
    for (int kk = 0; kk < 4; ++kk) {
#if USE_ASYNC_LDS
      v16h kf = frag_ld(Kt + (size_t)lr * HD_ + kk * 32 + 8 * half);
#else
      v16h kf = frag_ld(Kbase + (size_t)(kb + lr) * HD_ + kk * 32 + 8 * half);
#endif
      sa = __builtin_amdgcn_wmma_f32_16x16x32_f16(false, qf[kk], false, kf,
                                                  (short)0, sa, false, false);
    }
    if (do1) {
#pragma unroll
      for (int kk = 0; kk < 4; ++kk) {
#if USE_ASYNC_LDS
        v16h kf = frag_ld(Kt + (size_t)(16 + lr) * HD_ + kk * 32 + 8 * half);
#else
        v16h kf = frag_ld(Kbase + (size_t)(kb + 16 + lr) * HD_ + kk * 32 + 8 * half);
#endif
        sb = __builtin_amdgcn_wmma_f32_16x16x32_f16(false, qf[kk], false, kf,
                                                    (short)0, sb, false, false);
      }
    }

    // ---- causal mask + online softmax (row i+8*half spans 16 lanes of a half)
    float alpha[8];
#pragma unroll
    for (int i = 0; i < 8; ++i) {
      int row = qb + i + 8 * half;
      float s0 = sa[i] * scale;
      if (kb + lr > row) s0 = NEGINF;
      float s1 = do1 ? sb[i] * scale : NEGINF;
      if (kb + 16 + lr > row) s1 = NEGINF;

      float t = fmaxf(s0, s1);
#pragma unroll
      for (int msk = 8; msk >= 1; msk >>= 1) t = fmaxf(t, __shfl_xor(t, msk, 32));
      float mn = fmaxf(m[i], t);
      float al = __expf(m[i] - mn);
      float p0 = __expf(s0 - mn);
      float p1 = __expf(s1 - mn);
      float rs = p0 + p1;
#pragma unroll
      for (int msk = 8; msk >= 1; msk >>= 1) rs += __shfl_xor(rs, msk, 32);
      l[i] = l[i] * al + rs;
      m[i] = mn;
      alpha[i] = al;
      myP[(i + 8 * half) * 32 + lr]      = (_Float16)p0;   // P transpose via LDS
      myP[(i + 8 * half) * 32 + 16 + lr] = (_Float16)p1;
    }
#pragma unroll
    for (int t = 0; t < 8; ++t)
#pragma unroll
      for (int i = 0; i < 8; ++i) of[t][i] *= alpha[i];

    __builtin_amdgcn_wave_barrier();       // intra-wave: DS ops are in order
    v16h pa = frag_ld(&myP[lr * 32 + 8 * half]);   // P as A-frag (16x32 f16)
    __builtin_amdgcn_wave_barrier();

    // ---- O += P * V  (8 HD-tiles of 16)
#pragma unroll
    for (int nt = 0; nt < 8; ++nt) {
      v16h vb = frag_ld(Vbase + (size_t)(nt * 16 + lr) * S_ + kb + 8 * half);
      of[nt] = __builtin_amdgcn_wmma_f32_16x16x32_f16(false, pa, false, vb,
                                                      (short)0, of[nt], false, false);
    }

#if USE_ASYNC_LDS
    __builtin_amdgcn_s_wait_asynccnt(0);   // finish the prefetched tile
    cur ^= 1;
#endif
  }

  // ---- normalize + store token-major f16 [B*S, DQ] for the wo GEMM
  float inv[8];
#pragma unroll
  for (int i = 0; i < 8; ++i) inv[i] = 1.0f / l[i];
  _Float16* Obase = O + ((size_t)b * S_ + qb) * DQ_ + (size_t)h * HD_;
#pragma unroll
  for (int nt = 0; nt < 8; ++nt)
#pragma unroll
    for (int i = 0; i < 8; ++i)
      Obase[(size_t)(i + 8 * half) * DQ_ + nt * 16 + lr] = (_Float16)(of[nt][i] * inv[i]);
}

// ---------------------------------------------------------------- host
extern "C" void kernel_launch(void* const* d_in, const int* in_sizes, int n_in,
                              void* d_out, int out_size, void* d_ws, size_t ws_size,
                              hipStream_t stream) {
  const float* x  = (const float*)d_in[0];
  const float* fc = (const float*)d_in[1];
  const float* fs = (const float*)d_in[2];
  const float* wq = (const float*)d_in[3];
  const float* wk = (const float*)d_in[4];
  const float* wv = (const float*)d_in[5];
  const float* wo = (const float*)d_in[6];
  float* out = (float*)d_out;

  char* ws = (char*)d_ws;
  auto carve = [&](size_t bytes) {
    char* p = ws;
    ws += (bytes + 255) & ~(size_t)255;
    return p;
  };
  const long nX  = (long)B_ * S_ * E_;
  const long nWq = (long)DQ_ * E_;
  const long nWk = (long)DKV_ * E_;
  const long nWo = (long)E_ * DQ_;

  _Float16* xh  = (_Float16*)carve((size_t)nX * 2);
  _Float16* wqh = (_Float16*)carve((size_t)nWq * 2);
  _Float16* wkh = (_Float16*)carve((size_t)nWk * 2);
  _Float16* wvh = (_Float16*)carve((size_t)nWk * 2);
  _Float16* woh = (_Float16*)carve((size_t)nWo * 2);
  float* Qf = (float*)carve((size_t)B_ * S_ * DQ_ * 4);
  float* Kf = (float*)carve((size_t)B_ * S_ * DKV_ * 4);
  float* Vf = (float*)carve((size_t)B_ * S_ * DKV_ * 4);
  _Float16* Qh  = (_Float16*)carve((size_t)B_ * S_ * DQ_ * 2);
  _Float16* Kh  = (_Float16*)carve((size_t)B_ * S_ * DKV_ * 2);
  _Float16* Vth = (_Float16*)carve((size_t)B_ * S_ * DKV_ * 2);
  _Float16* Oh  = (_Float16*)Qf;   // reuse: Qf is dead after rope_pack(Q)

  const int T = 256;
  auto nb = [](long n, int t) { return (unsigned)((n + t - 1) / t); };

  cvt_f32_f16<<<dim3(4096), T, 0, stream>>>(x,  xh,  nX);
  cvt_f32_f16<<<dim3(4096), T, 0, stream>>>(wq, wqh, nWq);
  cvt_f32_f16<<<dim3(4096), T, 0, stream>>>(wk, wkh, nWk);
  cvt_f32_f16<<<dim3(4096), T, 0, stream>>>(wv, wvh, nWk);
  cvt_f32_f16<<<dim3(4096), T, 0, stream>>>(wo, woh, nWo);

  const int M = B_ * S_;  // 4096 tokens
  gemm_h_nt<<<dim3(DQ_ / 256, M / 64), T, 0, stream>>>(xh, wqh, Qf, M, DQ_, E_);
  gemm_h_nt<<<dim3(DKV_ / 256, M / 64), T, 0, stream>>>(xh, wkh, Kf, M, DKV_, E_);
  gemm_h_nt<<<dim3(DKV_ / 256, M / 64), T, 0, stream>>>(xh, wvh, Vf, M, DKV_, E_);

  const long nRq = (long)B_ * S_ * HQ_ * (HD_ / 2);
  const long nRk = (long)B_ * S_ * HKV_ * (HD_ / 2);
  const long nV  = (long)B_ * S_ * HKV_ * HD_;
  rope_pack<<<nb(nRq, T), T, 0, stream>>>(Qf, fc, fs, Qh, HQ_, nRq);
  rope_pack<<<nb(nRk, T), T, 0, stream>>>(Kf, fc, fs, Kh, HKV_, nRk);
  v_pack<<<nb(nV, T), T, 0, stream>>>(Vf, Vth, nV);

  flash_attn<<<dim3(S_ / 64, HQ_, B_), 128, 0, stream>>>(Qh, Kh, Vth, Oh);

  gemm_h_nt<<<dim3(E_ / 256, M / 64), T, 0, stream>>>(Oh, woh, out, M, E_, DQ_);
}
